// GATLayerWithRNNAndWeight_1932735283939
// MI455X (gfx1250) — matrix operand order repaired and checked by reading, not compiled, not run on hardware
//
#include <hip/hip_runtime.h>

// ---------------------------------------------------------------------------
// Types / helpers
// ---------------------------------------------------------------------------
typedef __attribute__((ext_vector_type(16))) __bf16 v16bf;
typedef __attribute__((ext_vector_type(8)))  __bf16 v8bf;
typedef __attribute__((ext_vector_type(8)))  float  v8f;

__device__ __forceinline__ __bf16 f2bf(float f) {
  union { float f; unsigned u; } in; in.f = f;
  unsigned u = in.u;
  unsigned r = u + 0x7FFFu + ((u >> 16) & 1u);   // round-to-nearest-even
  union { unsigned short s; __bf16 b; } out;
  out.s = (unsigned short)(r >> 16);
  return out.b;
}

__device__ __forceinline__ float leakyf(float x) { return x > 0.f ? x : 0.01f * x; }

__device__ __forceinline__ v8f v8zero() {
  v8f z;
#pragma unroll
  for (int i = 0; i < 8; ++i) z[i] = 0.f;
  return z;
}

__device__ __forceinline__ v8f wmma_bf16(v16bf a, v16bf b, v8f c) {
  return __builtin_amdgcn_wmma_f32_16x16x32_bf16(false, a, false, b, (short)0, c,
                                                 false, false);
}

// A fragment: 16x32 bf16 row-major, lane L: M=L&15, h=L>>4.
// Elements are two contiguous 8-elem runs -> two 16B vector loads.
__device__ __forceinline__ v16bf load_a_vec(const __bf16* __restrict__ A, int lda,
                                            int m0, int k0, int lane) {
  int m = lane & 15, hh = lane >> 4;
  const __bf16* rp = A + (size_t)(m0 + m) * lda + k0 + 16 * hh;
  v8bf lo = *(const v8bf*)rp;          // K = 16h .. 16h+7
  v8bf hi = *(const v8bf*)(rp + 8);    // K = 16h+8 .. 16h+15
  return __builtin_shufflevector(lo, hi, 0, 1, 2, 3, 4, 5, 6, 7, 8, 9, 10, 11, 12,
                                 13, 14, 15);
}

// Packed-B layout: for B[K x N], tile t = kt*(N/16)+ntn holds 32 lanes x 16 elems
// contiguous; lane L: N = ntn*16+(L&15), elem i: K = kt*32 + 16*(L>>4) + i.
__device__ __forceinline__ v16bf load_b_packed(const __bf16* __restrict__ Bp,
                                               int nTilesN, int kt, int ntn,
                                               int lane) {
  return *(const v16bf*)(Bp + (((size_t)(kt * nTilesN + ntn) * 32 + lane) << 4));
}

// ---------------------------------------------------------------------------
// Conversion / packing kernels
// ---------------------------------------------------------------------------
__global__ void cvt_f32_bf16(const float* __restrict__ src, __bf16* __restrict__ dst,
                             int n) {
  int i = blockIdx.x * blockDim.x + threadIdx.x;
  if (i < n) dst[i] = f2bf(src[i]);
}

// Pack f32 [K x N] (or transpose of [N x src_ld]) into fragment-ready bf16 tiles.
// K rows beyond Kvalid are zero-padded.
__global__ void pack_b_f32(const float* __restrict__ src, __bf16* __restrict__ dst,
                           int N, int Kvalid, int transpose, int src_ld, int total) {
  int idx = blockIdx.x * blockDim.x + threadIdx.x;
  if (idx >= total) return;
  int i = idx & 15;
  int lane = (idx >> 4) & 31;
  int t = idx >> 9;
  int nTilesN = N >> 4;
  int ntn = t % nTilesN, kt = t / nTilesN;
  int k = kt * 32 + 16 * (lane >> 4) + i;
  int n = ntn * 16 + (lane & 15);
  float v = 0.f;
  if (k < Kvalid)
    v = transpose ? src[(size_t)n * src_ld + k] : src[(size_t)k * src_ld + n];
  dst[idx] = f2bf(v);
}

// Repack row-major bf16 [Z x K x N] into per-batch fragment-ready tiles.
__global__ void pack_b_bf16_batched(const __bf16* __restrict__ src,
                                    __bf16* __restrict__ dst, int N, int perBatch,
                                    int total) {
  int idx = blockIdx.x * blockDim.x + threadIdx.x;
  if (idx >= total) return;
  int z = idx / perBatch;
  int r = idx - z * perBatch;
  int i = r & 15;
  int lane = (r >> 4) & 31;
  int t = r >> 9;
  int nTilesN = N >> 4;
  int ntn = t % nTilesN, kt = t / nTilesN;
  int k = kt * 32 + 16 * (lane >> 4) + i;
  int n = ntn * 16 + (lane & 15);
  dst[(size_t)z * perBatch + r] = src[(size_t)z * perBatch + (size_t)k * N + n];
}

// ---------------------------------------------------------------------------
// Generic batched bf16 WMMA GEMM: one wave computes a 16x64 C tile.
// grid = (M/16, N/64, Z).  B is in packed-tile layout.
// Software-pipelined K loop: prefetch k+1 fragments before issuing k's WMMAs.
// mode bit0: add bias[col]; bit1: ELU epilogue.
// ---------------------------------------------------------------------------
__global__ __launch_bounds__(32) void gemm_bf16_16x64(
    const __bf16* __restrict__ A, const __bf16* __restrict__ Bp,
    const float* __restrict__ bias, float* __restrict__ Cf, __bf16* __restrict__ Cb,
    int K, int lda, int nTilesN, int ldc, long sA, long sB, long sC, int mode) {
  int lane = threadIdx.x;
  int m0 = blockIdx.x * 16;
  int ntn0 = blockIdx.y * 4;
  int n0 = blockIdx.y * 64;
  int z = blockIdx.z;
  A += (size_t)z * sA;
  Bp += (size_t)z * sB;

  v8f acc[4];
#pragma unroll
  for (int nt = 0; nt < 4; ++nt) acc[nt] = v8zero();

  int nK = K >> 5;
  // prologue: fragments for kt = 0
  v16bf aCur = load_a_vec(A, lda, m0, 0, lane);
  v16bf bCur[4];
#pragma unroll
  for (int nt = 0; nt < 4; ++nt) bCur[nt] = load_b_packed(Bp, nTilesN, 0, ntn0 + nt, lane);

  for (int kt = 0; kt + 1 < nK; ++kt) {
    // prefetch kt+1 (waits for these land after the 4 WMMAs below)
    v16bf aNext = load_a_vec(A, lda, m0, (kt + 1) << 5, lane);
    v16bf b0 = load_b_packed(Bp, nTilesN, kt + 1, ntn0 + 0, lane);
    v16bf b1 = load_b_packed(Bp, nTilesN, kt + 1, ntn0 + 1, lane);
    v16bf b2 = load_b_packed(Bp, nTilesN, kt + 1, ntn0 + 2, lane);
    v16bf b3 = load_b_packed(Bp, nTilesN, kt + 1, ntn0 + 3, lane);
    acc[0] = wmma_bf16(aCur, bCur[0], acc[0]);
    acc[1] = wmma_bf16(aCur, bCur[1], acc[1]);
    acc[2] = wmma_bf16(aCur, bCur[2], acc[2]);
    acc[3] = wmma_bf16(aCur, bCur[3], acc[3]);
    aCur = aNext;
    bCur[0] = b0; bCur[1] = b1; bCur[2] = b2; bCur[3] = b3;
  }
  // epilogue step
  acc[0] = wmma_bf16(aCur, bCur[0], acc[0]);
  acc[1] = wmma_bf16(aCur, bCur[1], acc[1]);
  acc[2] = wmma_bf16(aCur, bCur[2], acc[2]);
  acc[3] = wmma_bf16(aCur, bCur[3], acc[3]);

  int n = lane & 15, hh = lane >> 4;
#pragma unroll
  for (int nt = 0; nt < 4; ++nt) {
#pragma unroll
    for (int v = 0; v < 8; ++v) {
      int row = m0 + v + 8 * hh;
      int col = n0 + nt * 16 + n;
      float val = acc[nt][v];
      if (mode & 1) val += bias[col];
      if (mode & 2) val = val > 0.f ? val : (expf(val) - 1.f);
      size_t off = (size_t)z * sC + (size_t)row * ldc + col;
      if (Cf) Cf[off] = val;
      if (Cb) Cb[off] = f2bf(val);
    }
  }
}

// ---------------------------------------------------------------------------
// s[row] = dot(H[row, 0:256], w)  (one wave per row, 8 rows per block)
// ---------------------------------------------------------------------------
__global__ __launch_bounds__(256) void row_dot256(const float* __restrict__ Hm,
                                                  const float* __restrict__ w,
                                                  float* __restrict__ out, int nrows) {
  int wv = threadIdx.x >> 5, lane = threadIdx.x & 31;
  int row = blockIdx.x * 8 + wv;
  if (row >= nrows) return;
  const float* rp = Hm + (size_t)row * 256;
  float p = 0.f;
#pragma unroll
  for (int k = lane; k < 256; k += 32) p += rp[k] * w[k];
#pragma unroll
  for (int off = 16; off; off >>= 1) p += __shfl_xor(p, off, 32);
  if (lane == 0) out[row] = p;
}

// ---------------------------------------------------------------------------
// Fused edge-score + masked softmax.  One block per (b,i) row:
//   wproj = w_edge[b,i,:, :14] @ wfc_w (K padded to 32, via WMMA, LDS-staged A)
//   e[j] = sum_k leaky(wproj+wfc_b)*fc_w[64+k] + sum_k leaky(e1_i+e2_j)*fc_w[k] + fc_b
//   att = softmax_j(where(adj>0, e, -9e15))
// ---------------------------------------------------------------------------
__global__ __launch_bounds__(256) void edge_softmax(
    const float* __restrict__ w_edge, const __bf16* __restrict__ wfcp,
    const float* __restrict__ wfc_bias, const float* __restrict__ fc_w,
    const float* __restrict__ fc_b, const float* __restrict__ e1,
    const float* __restrict__ e2, const int* __restrict__ adj,
    float* __restrict__ att, __bf16* __restrict__ attb) {
  __shared__ __bf16 wedge[128 * 32];  // per-row K padded 14 -> 32
  __shared__ float erow[128];
  __shared__ float red[128];
  int bi = blockIdx.x;  // b*128 + i
  int tid = threadIdx.x;
  int wv = tid >> 5, lane = tid & 31;
  int m = lane & 15, hh = lane >> 4;
  int j0 = wv * 16;

  // --- stage w_edge[b,i,:,:] (contiguous 128*14 floats) into padded bf16 LDS ---
  const float* wsrc = w_edge + (size_t)bi * 128 * 14;
  for (int i = tid; i < 128 * 32; i += 256) {
    int row = i >> 5, k = i & 31;
    wedge[i] = (k < 14) ? f2bf(wsrc[row * 14 + k]) : f2bf(0.f);
  }
  __syncthreads();

  // --- A fragment from LDS: two ds_load_b128 per lane ---
  v16bf afrag;
  {
    const __bf16* rp = &wedge[(j0 + m) * 32 + 16 * hh];
    v8bf lo = *(const v8bf*)rp;
    v8bf hi = *(const v8bf*)(rp + 8);
    afrag = __builtin_shufflevector(lo, hi, 0, 1, 2, 3, 4, 5, 6, 7, 8, 9, 10, 11, 12,
                                    13, 14, 15);
  }

  float part[8];
#pragma unroll
  for (int v = 0; v < 8; ++v) part[v] = 0.f;

#pragma unroll
  for (int nt = 0; nt < 4; ++nt) {
    v16bf bfrag = load_b_packed(wfcp, 4, 0, nt, lane);
    v8f accw = wmma_bf16(afrag, bfrag, v8zero());
    int col = nt * 16 + m;
    float wb = wfc_bias[col], fw = fc_w[64 + col];
#pragma unroll
    for (int v = 0; v < 8; ++v) part[v] += leakyf(accw[v] + wb) * fw;
  }
  // reduce over the 16 lanes of each half (rows j0+v+8*hh)
#pragma unroll
  for (int v = 0; v < 8; ++v)
#pragma unroll
    for (int off = 1; off <= 8; off <<= 1) part[v] += __shfl_xor(part[v], off, 32);
  if (m == 0) {
#pragma unroll
    for (int v = 0; v < 8; ++v) erow[j0 + v + 8 * hh] = part[v];
  }

  // --- e1_i + e2_j contribution ---
  {
    int j = j0 + m;
    const float* r1 = e1 + (size_t)bi * 64;
    const float* r2 = e2 + (((size_t)(bi >> 7)) * 128 + j) * 64;
    float p = 0.f;
#pragma unroll
    for (int k = 32 * hh; k < 32 * hh + 32; ++k) p += leakyf(r1[k] + r2[k]) * fc_w[k];
    p += __shfl_xor(p, 16, 32);
    if (hh == 0) erow[j] += p + fc_b[0];
  }
  __syncthreads();

  // --- masked softmax over 128 entries ---
  float ev = 0.f, ex = 0.f;
  if (tid < 128) {
    ev = erow[tid];
    if (adj[(size_t)bi * 128 + tid] <= 0) ev = -9e15f;
    red[tid] = ev;
  }
  __syncthreads();
  for (int s = 64; s > 0; s >>= 1) {
    if (tid < s) red[tid] = fmaxf(red[tid], red[tid + s]);
    __syncthreads();
  }
  float mx = red[0];
  __syncthreads();
  if (tid < 128) {
    ex = expf(ev - mx);
    red[tid] = ex;
  }
  __syncthreads();
  for (int s = 64; s > 0; s >>= 1) {
    if (tid < s) red[tid] += red[tid + s];
    __syncthreads();
  }
  float tot = red[0];
  if (tid < 128) {
    float av = ex / tot;
    att[(size_t)bi * 128 + tid] = av;
    attb[(size_t)bi * 128 + tid] = f2bf(av);
  }
}

// ---------------------------------------------------------------------------
// hn0[b,i,j] = leaky(att * s[b,i] + fcc_b)   (RNN input, bf16)
// ---------------------------------------------------------------------------
__global__ void hn0_kernel(const float* __restrict__ att, const float* __restrict__ s,
                           const float* __restrict__ fcc_b, __bf16* __restrict__ hnb,
                           int n) {
  int i = blockIdx.x * blockDim.x + threadIdx.x;
  if (i < n) {
    float v = att[i] * s[i >> 7] + fcc_b[0];
    hnb[i] = f2bf(leakyf(v));
  }
}

// ---------------------------------------------------------------------------
// Bidirectional RNN scan: block 0 = forward, block 1 = backward.
// state [16 batches x 64 hidden] in LDS, whh in LDS, 128 serial steps.
// Writes leaky(h_t) into the concat buffer hx [2048 x 384] at cols dir*64..+63.
// ---------------------------------------------------------------------------
__global__ __launch_bounds__(256) void rnn_scan(
    const float* __restrict__ xwf, const float* __restrict__ xwb,
    const float* __restrict__ whh_f, const float* __restrict__ whh_b,
    const float* __restrict__ bhh_f, const float* __restrict__ bhh_b,
    __bf16* __restrict__ hx) {
  __shared__ float Wl[64 * 64];
  __shared__ float st[16 * 64];
  int dir = blockIdx.x;
  const float* xw = dir ? xwb : xwf;
  const float* wh = dir ? whh_b : whh_f;
  const float* bh = dir ? bhh_b : bhh_f;
  int tid = threadIdx.x;
  for (int i = tid; i < 4096; i += 256) Wl[i] = wh[i];
  for (int i = tid; i < 1024; i += 256) st[i] = 0.f;
  __syncthreads();

  int b = tid >> 4;          // 0..15 batch
  int c0 = (tid & 15) * 4;   // 4 hidden channels per thread

  for (int t = 0; t < 128; ++t) {
    int ti = dir ? (127 - t) : t;  // input (and output) time index
    const float* xr = xw + ((size_t)b * 128 + ti) * 64;
    float acc[4];
#pragma unroll
    for (int q = 0; q < 4; ++q) acc[q] = xr[c0 + q] + bh[c0 + q];
    const float* sb = st + b * 64;
    for (int k = 0; k < 64; ++k) {
      float hv = sb[k];
#pragma unroll
      for (int q = 0; q < 4; ++q) acc[q] += hv * Wl[(c0 + q) * 64 + k];
    }
    __syncthreads();
#pragma unroll
    for (int q = 0; q < 4; ++q) {
      float h = tanhf(acc[q]);
      st[b * 64 + c0 + q] = h;
      hx[((size_t)b * 128 + ti) * 384 + dir * 64 + c0 + q] = f2bf(leakyf(h));
    }
    __syncthreads();
  }
}

// ---------------------------------------------------------------------------
// Launch
// ---------------------------------------------------------------------------
extern "C" void kernel_launch(void* const* d_in, const int* in_sizes, int n_in,
                              void* d_out, int out_size, void* d_ws, size_t ws_size,
                              hipStream_t stream) {
  (void)in_sizes; (void)n_in; (void)out_size; (void)ws_size;
  const float* x      = (const float*)d_in[0];
  const int*   adj    = (const int*)d_in[1];
  const float* w_edge = (const float*)d_in[2];
  const float* W      = (const float*)d_in[3];
  const float* a      = (const float*)d_in[4];
  const float* wfc_w  = (const float*)d_in[5];
  const float* wfc_b  = (const float*)d_in[6];
  const float* fc_w   = (const float*)d_in[7];
  const float* fc_b   = (const float*)d_in[8];
  const float* fcc_w  = (const float*)d_in[9];
  const float* fcc_b  = (const float*)d_in[10];
  const float* wih_f  = (const float*)d_in[11];
  const float* whh_f  = (const float*)d_in[12];
  const float* bih_f  = (const float*)d_in[13];
  const float* bhh_f  = (const float*)d_in[14];
  const float* wih_b  = (const float*)d_in[15];
  const float* whh_b  = (const float*)d_in[16];
  const float* bih_b  = (const float*)d_in[17];
  const float* bhh_b  = (const float*)d_in[18];
  const float* fco_w  = (const float*)d_in[19];
  const float* fco_b  = (const float*)d_in[20];
  float* out = (float*)d_out;

  char* ws = (char*)d_ws;
  size_t off = 0;
  auto alloc = [&](size_t bytes) -> void* {
    void* p = ws + off;
    off = (off + bytes + 255) & ~(size_t)255;
    return p;
  };

  // bf16 buffers (all B operands in packed-tile layout)
  __bf16* xbf   = (__bf16*)alloc(262144 * 2);      // A: [16,128,128]
  __bf16* Wp    = (__bf16*)alloc(32768 * 2);       // B: [128,256] packed
  __bf16* aTp   = (__bf16*)alloc(16384 * 2);       // B: a_top [256,64] packed
  __bf16* aBp   = (__bf16*)alloc(16384 * 2);       // B: a_bot [256,64] packed
  __bf16* wfcp  = (__bf16*)alloc(32 * 64 * 2);     // B: [32,64] packed (K pad 14)
  __bf16* wihfp = (__bf16*)alloc(8192 * 2);        // B: wih_f^T [128,64] packed
  __bf16* wihbp = (__bf16*)alloc(8192 * 2);        // B: wih_b^T [128,64] packed
  __bf16* fcop  = (__bf16*)alloc(98304 * 2);       // B: [384,256] packed
  float*  h     = (float*)alloc(524288 * 4);       // [16,128,256]
  __bf16* hbf   = (__bf16*)alloc(524288 * 2);      // A: row-major
  __bf16* hbp   = (__bf16*)alloc(524288 * 2);      // B: packed per batch
  float*  e1    = (float*)alloc(131072 * 4);       // [2048,64]
  float*  e2    = (float*)alloc(131072 * 4);
  float*  s     = (float*)alloc(2048 * 4);
  float*  att   = (float*)alloc(262144 * 4);       // [16,128,128]
  __bf16* attbf = (__bf16*)alloc(262144 * 2);
  __bf16* hnbf  = (__bf16*)alloc(262144 * 2);      // RNN input
  float*  xwf   = (float*)alloc(131072 * 4);       // [2048,64]
  float*  xwb   = (float*)alloc(131072 * 4);
  __bf16* hx    = (__bf16*)alloc(2048 * 384 * 2);  // concat [hn_rnn | h1]

  // --- conversions / packing ---
  cvt_f32_bf16<<<1024, 256, 0, stream>>>(x, xbf, 262144);
  pack_b_f32<<<128, 256, 0, stream>>>(W, Wp, 256, 128, 0, 256, 32768);
  pack_b_f32<<<64, 256, 0, stream>>>(a, aTp, 64, 256, 0, 64, 16384);
  pack_b_f32<<<64, 256, 0, stream>>>(a + 256 * 64, aBp, 64, 256, 0, 64, 16384);
  pack_b_f32<<<8, 256, 0, stream>>>(wfc_w, wfcp, 64, 14, 0, 64, 2048);
  pack_b_f32<<<32, 256, 0, stream>>>(wih_f, wihfp, 64, 128, 1, 128, 8192);
  pack_b_f32<<<32, 256, 0, stream>>>(wih_b, wihbp, 64, 128, 1, 128, 8192);
  pack_b_f32<<<384, 256, 0, stream>>>(fco_w, fcop, 256, 384, 0, 256, 98304);

  // --- h = x @ W  (per batch 128x256, K=128) ---
  gemm_bf16_16x64<<<dim3(8, 4, 16), 32, 0, stream>>>(
      xbf, Wp, nullptr, h, hbf, 128, 128, 16, 256, 16384, 0, 32768, 0);

  // repack h as packed-B for the att@h GEMM
  pack_b_bf16_batched<<<2048, 256, 0, stream>>>(hbf, hbp, 256, 32768, 524288);

  // --- e1 = h @ a_top, e2 = h @ a_bot  (M=2048, N=64, K=256) ---
  gemm_bf16_16x64<<<dim3(128, 1, 1), 32, 0, stream>>>(
      hbf, aTp, nullptr, e1, nullptr, 256, 256, 4, 64, 0, 0, 0, 0);
  gemm_bf16_16x64<<<dim3(128, 1, 1), 32, 0, stream>>>(
      hbf, aBp, nullptr, e2, nullptr, 256, 256, 4, 64, 0, 0, 0, 0);

  // --- s = h . fcc_w ---
  row_dot256<<<256, 256, 0, stream>>>(h, fcc_w, s, 2048);

  // --- edge scores + masked softmax ---
  edge_softmax<<<2048, 256, 0, stream>>>(w_edge, wfcp, wfc_b, fc_w, fc_b, e1, e2,
                                         adj, att, attbf);

  // --- RNN input ---
  hn0_kernel<<<1024, 256, 0, stream>>>(att, s, fcc_b, hnbf, 262144);

  // --- xw_f / xw_b = hn @ wih^T + bih  (M=2048, N=64, K=128) ---
  gemm_bf16_16x64<<<dim3(128, 1, 1), 32, 0, stream>>>(
      hnbf, wihfp, bih_f, xwf, nullptr, 128, 128, 4, 64, 0, 0, 0, 1);
  gemm_bf16_16x64<<<dim3(128, 1, 1), 32, 0, stream>>>(
      hnbf, wihbp, bih_b, xwb, nullptr, 128, 128, 4, 64, 0, 0, 0, 1);

  // --- h1 = att @ h  -> concat buffer cols 128..383 ---
  gemm_bf16_16x64<<<dim3(8, 16, 16), 32, 0, stream>>>(
      attbf, hbp, nullptr, nullptr, hx + 128, 128, 128, 16, 384, 16384, 32768,
      128 * 384, 0);

  // --- bidirectional RNN scan -> concat buffer cols 0..127 ---
  rnn_scan<<<2, 256, 0, stream>>>(xwf, xwb, whh_f, whh_b, bhh_f, bhh_b, hx);

  // --- out = elu(concat @ fco_w + fco_b)  (M=2048, N=256, K=384) ---
  gemm_bf16_16x64<<<dim3(128, 4, 1), 32, 0, stream>>>(
      hx, fcop, fco_b, out, nullptr, 384, 384, 16, 256, 0, 0, 0, 3);
}
